// SuperRegionGraphGenerator_82755429859612
// MI455X (gfx1250) — matrix operand order repaired and checked by reading, not compiled, not run on hardware
//
#include <hip/hip_runtime.h>

// ---------------------------------------------------------------------------
// SuperRegionGraphGenerator for MI455X (gfx1250, wave32, WMMA).
//
// Pipeline (all on `stream`):
//   1) pack_x:     feature_map [8,256,96,96] + 5x coords -> X f16 [8,9216,288]
//   2) pack_w:     W_enc f32 [128,259] -> Wh f16 [128,288]
//   3) init_cent:  strided-point init of 128 centroids/batch (+ ||c||^2)
//   4) 10x { zero accum; assign (WMMA f16 GEMM + in-register argmin +
//            atomic scatter-add); update (sums/counts -> centroids, ||c||^2) }
//   5) feats:      centroids @ W_enc^T + b_enc via WMMA  -> d_out[0 .. 8*128*128)
//   6) adj:        8-NN symmetric adjacency on coord dims -> d_out[8*128*128 ..)
//
// Workspace: ~45.5 MB (X fits in the 192MB L2, so the 10 assignment GEMM
// passes are L2-resident and compute-bound on v_wmma_f32_16x16x32_f16).
// ---------------------------------------------------------------------------

typedef __attribute__((ext_vector_type(16))) _Float16 v16h;
typedef __attribute__((ext_vector_type(8)))  float    v8f;

#define BATCH 8
#define CCH   256
#define HH    96
#define WW    96
#define NPT   (HH * WW)   // 9216 points per batch
#define DDIM  259         // 256 features + 3 coords
#define DPAD  288         // padded to 9 * 32 for the WMMA K-loop
#define KCL   128         // clusters
#define FOUT  128         // encoder output features
#define NITER 10

// ---------------- pack feature map + coords into f16 X ---------------------
__global__ __launch_bounds__(256) void pack_x_kernel(
    const float* __restrict__ fm, _Float16* __restrict__ X)
{
    const int b = blockIdx.y;
    const int n = blockIdx.x * 256 + threadIdx.x;   // pixel index
    const int h = n / WW, w = n % WW;
    _Float16* xr = X + ((size_t)b * NPT + n) * DPAD;
    const float* fb = fm + (size_t)b * CCH * NPT + n;
    for (int c = 0; c < CCH; ++c)
        xr[c] = (_Float16)fb[(size_t)c * NPT];
    xr[256] = (_Float16)(0.5f * 5.0f);                        // gz * 5
    xr[257] = (_Float16)((float)h / (float)(HH - 1) * 5.0f);  // gy * 5
    xr[258] = (_Float16)((float)w / (float)(WW - 1) * 5.0f);  // gx * 5
    for (int c = DDIM; c < DPAD; ++c) xr[c] = (_Float16)0.0f; // zero pad
}

// ---------------- pack W_enc into padded f16 -------------------------------
__global__ void pack_w_kernel(const float* __restrict__ Wenc,
                              _Float16* __restrict__ Wh)
{
    const int f = blockIdx.x;   // 0..127
    const int c = threadIdx.x;  // 0..287
    Wh[(size_t)f * DPAD + c] =
        (c < DDIM) ? (_Float16)Wenc[(size_t)f * DDIM + c] : (_Float16)0.0f;
}

// ---------------- initial centroids: strided points ------------------------
__global__ __launch_bounds__(128) void init_cent_kernel(
    const _Float16* __restrict__ X, _Float16* __restrict__ Cent,
    float* __restrict__ Cent32, float* __restrict__ cnorm)
{
    const int k = blockIdx.x, b = blockIdx.y, tid = threadIdx.x;
    const int n = k * (NPT / KCL);
    const _Float16* xr = X + ((size_t)b * NPT + n) * DPAD;
    float sq = 0.f;
    for (int c = tid; c < DPAD; c += 128) {
        float v = (float)xr[c];
        Cent  [((size_t)b * KCL + k) * DPAD + c] = (_Float16)v;
        Cent32[((size_t)b * KCL + k) * DPAD + c] = v;
        sq += v * v;
    }
    __shared__ float red[128];
    red[tid] = sq; __syncthreads();
    for (int s = 64; s > 0; s >>= 1) {
        if (tid < s) red[tid] += red[tid + s];
        __syncthreads();
    }
    if (tid == 0) cnorm[b * KCL + k] = red[0];
}

// ---------------- zero accumulators ----------------------------------------
__global__ void zero_kernel(float* __restrict__ sums, float* __restrict__ counts)
{
    const int idx = blockIdx.x * 256 + threadIdx.x;
    if (idx < BATCH * KCL * DPAD) sums[idx] = 0.f;
    if (idx < BATCH * KCL)        counts[idx] = 0.f;
}

// ---------------- K-means assignment: WMMA GEMM + argmin + scatter ---------
// One block = 256 threads = 8 wave32s, handles a 128-point x 128-cluster tile.
// Wave w computes rows w*16..w*16+15 across all 128 clusters (8 WMMA tiles).
__global__ __launch_bounds__(256) void assign_kernel(
    const _Float16* __restrict__ X, const _Float16* __restrict__ Cent,
    const float* __restrict__ cnorm, float* __restrict__ sums,
    float* __restrict__ counts)
{
    const int b    = blockIdx.y;
    const int tile = blockIdx.x;            // 72 tiles of 128 points
    const int tid  = threadIdx.x;
    const int w    = tid >> 5;
    const int lane = tid & 31;

    __shared__ __align__(16) _Float16 lA[128 * 32];  // point panel
    __shared__ __align__(16) _Float16 lB[128 * 32];  // centroid panel
    __shared__ float lCn[KCL];
    __shared__ int   lLab[128];

    if (tid < KCL) lCn[tid] = cnorm[b * KCL + tid];

    const _Float16* Xb = X    + ((size_t)b * NPT + (size_t)tile * 128) * DPAD;
    const _Float16* Cb = Cent + (size_t)b * KCL * DPAD;

    v8f acc[8];
    #pragma unroll
    for (int t = 0; t < 8; ++t)
        #pragma unroll
        for (int e = 0; e < 8; ++e) acc[t][e] = 0.f;

    for (int k0 = 0; k0 < DPAD; k0 += 32) {
        __syncthreads();
        // Stage 128x32 panels of A (points) and B (centroids): 512 16B chunks
        #pragma unroll
        for (int i = 0; i < 2; ++i) {
            const int id = tid + i * 256;        // 0..511
            const int r  = id >> 2;              // row 0..127
            const int q  = id & 3;               // 8-half chunk 0..3
            const float4* sa =
                reinterpret_cast<const float4*>(Xb + (size_t)r * DPAD + k0 + q * 8);
            *reinterpret_cast<float4*>(&lA[r * 32 + q * 8]) = *sa;
            const float4* sb =
                reinterpret_cast<const float4*>(Cb + (size_t)r * DPAD + k0 + q * 8);
            *reinterpret_cast<float4*>(&lB[r * 32 + q * 8]) = *sb;
        }
        __syncthreads();

        // A fragment (16x32 f16): lanes 0-15 hold K {0..7,16..23}, lanes
        // 16-31 hold K {8..15,24..31}, row = w*16 + lane%16.
        const int row  = w * 16 + (lane & 15);
        const int koff = (lane < 16) ? 0 : 8;
        v16h a;
        #pragma unroll
        for (int i = 0; i < 8; ++i) {
            a[i]     = lA[row * 32 + koff + i];
            a[8 + i] = lA[row * 32 + 16 + koff + i];
        }
        #pragma unroll
        for (int t = 0; t < 8; ++t) {
            // B fragment (32x16): col = lane%16, lanes 0-15 K {0..7,16..23}, lanes 16-31 K {8..15,24..31}.
            const int col = t * 16 + (lane & 15);
            const int kb  = (lane < 16) ? 0 : 8;
            v16h bf;
            #pragma unroll
            for (int i = 0; i < 16; ++i) bf[i] = lB[col * 32 + kb + (i & 7) + ((i >> 3) << 4)];
            acc[t] = __builtin_amdgcn_wmma_f32_16x16x32_f16(
                false, a, false, bf, (short)0, acc[t], false, false);
        }
    }

    // In-register argmin over clusters: score = ||c||^2 - 2*(x.c).
    // C layout: lane l, VGPR j -> row = j + (l>=16 ? 8 : 0), col = l%16.
    #pragma unroll
    for (int j = 0; j < 8; ++j) {
        float best = 3.4e38f; int bestk = 0;
        #pragma unroll
        for (int t = 0; t < 8; ++t) {
            const int col = t * 16 + (lane & 15);
            const float s = lCn[col] - 2.0f * acc[t][j];
            if (s < best) { best = s; bestk = col; }
        }
        #pragma unroll
        for (int off = 1; off < 16; off <<= 1) {   // reduce each 16-lane half
            const float ob = __shfl_xor(best, off, 32);
            const int   ok = __shfl_xor(bestk, off, 32);
            if (ob < best || (ob == best && ok < bestk)) { best = ob; bestk = ok; }
        }
        if (lane == 0)  lLab[w * 16 + j]     = bestk;
        if (lane == 16) lLab[w * 16 + j + 8] = bestk;
    }
    __syncthreads();

    if (tid < 128) atomicAdd(&counts[b * KCL + lLab[tid]], 1.0f);
    for (int r = 0; r < 128; ++r) {
        const int lab = lLab[r];
        const _Float16* xr = Xb + (size_t)r * DPAD;
        float* sr = sums + ((size_t)b * KCL + lab) * DPAD;
        for (int c = tid; c < DPAD; c += 256)
            atomicAdd(&sr[c], (float)xr[c]);
    }
}

// ---------------- centroid update + ||c||^2 --------------------------------
__global__ __launch_bounds__(128) void update_kernel(
    const float* __restrict__ sums, const float* __restrict__ counts,
    _Float16* __restrict__ Cent, float* __restrict__ Cent32,
    float* __restrict__ cnorm)
{
    const int k = blockIdx.x, b = blockIdx.y, tid = threadIdx.x;
    float cnt = counts[b * KCL + k];
    if (cnt < 1.f) cnt = 1.f;
    const float inv = 1.f / cnt;
    float sq = 0.f;
    const float* sr = sums + ((size_t)b * KCL + k) * DPAD;
    for (int c = tid; c < DPAD; c += 128) {
        const float v = sr[c] * inv;
        Cent  [((size_t)b * KCL + k) * DPAD + c] = (_Float16)v;
        Cent32[((size_t)b * KCL + k) * DPAD + c] = v;
        sq += v * v;
    }
    __shared__ float red[128];
    red[tid] = sq; __syncthreads();
    for (int s = 64; s > 0; s >>= 1) {
        if (tid < s) red[tid] += red[tid + s];
        __syncthreads();
    }
    if (tid == 0) cnorm[b * KCL + k] = red[0];
}

// ---------------- feats = centroids @ W_enc^T + b_enc (WMMA) ---------------
__global__ __launch_bounds__(256) void feats_kernel(
    const _Float16* __restrict__ Cent, const _Float16* __restrict__ Wh,
    const float* __restrict__ benc, float* __restrict__ out)
{
    const int b    = blockIdx.x;
    const int tid  = threadIdx.x;
    const int w    = tid >> 5;
    const int lane = tid & 31;

    __shared__ __align__(16) _Float16 lA[128 * 32];
    __shared__ __align__(16) _Float16 lB[128 * 32];

    const _Float16* Ab = Cent + (size_t)b * KCL * DPAD;

    v8f acc[8];
    #pragma unroll
    for (int t = 0; t < 8; ++t)
        #pragma unroll
        for (int e = 0; e < 8; ++e) acc[t][e] = 0.f;

    for (int k0 = 0; k0 < DPAD; k0 += 32) {
        __syncthreads();
        #pragma unroll
        for (int i = 0; i < 2; ++i) {
            const int id = tid + i * 256;
            const int r  = id >> 2;
            const int q  = id & 3;
            const float4* sa =
                reinterpret_cast<const float4*>(Ab + (size_t)r * DPAD + k0 + q * 8);
            *reinterpret_cast<float4*>(&lA[r * 32 + q * 8]) = *sa;
            const float4* sb =
                reinterpret_cast<const float4*>(Wh + (size_t)r * DPAD + k0 + q * 8);
            *reinterpret_cast<float4*>(&lB[r * 32 + q * 8]) = *sb;
        }
        __syncthreads();

        const int row  = w * 16 + (lane & 15);
        const int koff = (lane < 16) ? 0 : 8;
        v16h a;
        #pragma unroll
        for (int i = 0; i < 8; ++i) {
            a[i]     = lA[row * 32 + koff + i];
            a[8 + i] = lA[row * 32 + 16 + koff + i];
        }
        #pragma unroll
        for (int t = 0; t < 8; ++t) {
            const int col = t * 16 + (lane & 15);
            const int kb  = (lane < 16) ? 0 : 8;
            v16h bf;
            #pragma unroll
            for (int i = 0; i < 16; ++i) bf[i] = lB[col * 32 + kb + (i & 7) + ((i >> 3) << 4)];
            acc[t] = __builtin_amdgcn_wmma_f32_16x16x32_f16(
                false, a, false, bf, (short)0, acc[t], false, false);
        }
    }

    #pragma unroll
    for (int t = 0; t < 8; ++t) {
        const int col  = t * 16 + (lane & 15);
        const float bias = benc[col];
        #pragma unroll
        for (int j = 0; j < 8; ++j) {
            const int row = w * 16 + j + ((lane < 16) ? 0 : 8);
            out[((size_t)b * KCL + row) * FOUT + col] = acc[t][j] + bias;
        }
    }
}

// ---------------- 8-NN symmetric adjacency on coord dims -------------------
__global__ __launch_bounds__(128) void adj_kernel(
    const float* __restrict__ Cent32, float* __restrict__ outAdj)
{
    const int b = blockIdx.x;
    const int i = threadIdx.x;   // centroid row

    __shared__ float cz[KCL], cy[KCL], cx[KCL];
    __shared__ unsigned sel[KCL][4];

    const float* cr = Cent32 + ((size_t)b * KCL + i) * DPAD;
    cz[i] = cr[256]; cy[i] = cr[257]; cx[i] = cr[258];
    __syncthreads();

    const float z0 = cz[i], y0 = cy[i], x0 = cx[i];
    unsigned m0 = 0, m1 = 0, m2 = 0, m3 = 0;
    for (int pass = 0; pass < 8; ++pass) {   // 8 nearest (self included, d=0)
        float best = 3.4e38f; int bj = 0;
        for (int j = 0; j < KCL; ++j) {
            const unsigned bit = (j < 32) ? (m0 >> j) :
                                 (j < 64) ? (m1 >> (j - 32)) :
                                 (j < 96) ? (m2 >> (j - 64)) : (m3 >> (j - 96));
            if (bit & 1u) continue;
            const float dz = cz[j] - z0, dy = cy[j] - y0, dx = cx[j] - x0;
            const float d2 = dz * dz + dy * dy + dx * dx;
            if (d2 < best) { best = d2; bj = j; }
        }
        if (bj < 32)      m0 |= 1u << bj;
        else if (bj < 64) m1 |= 1u << (bj - 32);
        else if (bj < 96) m2 |= 1u << (bj - 64);
        else              m3 |= 1u << (bj - 96);
    }
    sel[i][0] = m0; sel[i][1] = m1; sel[i][2] = m2; sel[i][3] = m3;
    __syncthreads();

    float* row = outAdj + ((size_t)b * KCL + i) * KCL;
    for (int j = 0; j < KCL; ++j) {
        const unsigned fwd = (sel[i][j >> 5] >> (j & 31)) & 1u;
        const unsigned bwd = (sel[j][i >> 5] >> (i & 31)) & 1u;
        row[j] = (fwd | bwd) ? 1.f : 0.f;
    }
}

// ---------------------------------------------------------------------------
extern "C" void kernel_launch(void* const* d_in, const int* in_sizes, int n_in,
                              void* d_out, int out_size, void* d_ws, size_t ws_size,
                              hipStream_t stream)
{
    (void)in_sizes; (void)n_in; (void)out_size; (void)ws_size;
    const float* fm   = (const float*)d_in[0];  // [8,256,1,96,96]
    const float* Wenc = (const float*)d_in[1];  // [128,259]
    const float* benc = (const float*)d_in[2];  // [128]
    float* out = (float*)d_out;                 // feats [8,128,128] ++ adj [8,128,128]

    char* ws = (char*)d_ws;
    size_t off = 0;
    _Float16* X      = (_Float16*)(ws + off); off += (size_t)BATCH * NPT * DPAD * 2; // 42.5 MB
    _Float16* Cent   = (_Float16*)(ws + off); off += (size_t)BATCH * KCL * DPAD * 2;
    float*    Cent32 = (float*)   (ws + off); off += (size_t)BATCH * KCL * DPAD * 4;
    float*    sums   = (float*)   (ws + off); off += (size_t)BATCH * KCL * DPAD * 4;
    float*    counts = (float*)   (ws + off); off += (size_t)BATCH * KCL * 4;
    float*    cnorm  = (float*)   (ws + off); off += (size_t)BATCH * KCL * 4;
    _Float16* Wh     = (_Float16*)(ws + off); off += (size_t)FOUT * DPAD * 2;
    // total ~45.5 MB

    pack_x_kernel<<<dim3(NPT / 256, BATCH), 256, 0, stream>>>(fm, X);
    pack_w_kernel<<<dim3(FOUT), dim3(DPAD), 0, stream>>>(Wenc, Wh);
    init_cent_kernel<<<dim3(KCL, BATCH), 128, 0, stream>>>(X, Cent, Cent32, cnorm);

    for (int it = 0; it < NITER; ++it) {
        zero_kernel<<<dim3((BATCH * KCL * DPAD + 255) / 256), 256, 0, stream>>>(sums, counts);
        assign_kernel<<<dim3(NPT / 128, BATCH), 256, 0, stream>>>(X, Cent, cnorm, sums, counts);
        update_kernel<<<dim3(KCL, BATCH), 128, 0, stream>>>(sums, counts, Cent, Cent32, cnorm);
    }

    feats_kernel<<<dim3(BATCH), 256, 0, stream>>>(Cent, Wh, benc, out);
    adj_kernel<<<dim3(BATCH), 128, 0, stream>>>(Cent32, out + (size_t)BATCH * KCL * FOUT);
}
